// ReverseGRU_46926812677049
// MI455X (gfx1250) — matrix operand order repaired and checked by reading, not compiled, not run on hardware
//
#include <hip/hip_runtime.h>

// ---------------------------------------------------------------------------
// CDNA5 (gfx1250) implementation of the ReverseGRU + SplineConv forward pass.
// wave32; matrix work uses v_wmma_f32_16x16x32_f16 (f16 in, f32 accumulate).
// Spline weights staged block-wide into LDS via GLOBAL_LOAD_ASYNC_TO_LDS_B128
// (ASYNCcnt) so each 8-wave workgroup reads them once.
// ---------------------------------------------------------------------------

typedef __attribute__((ext_vector_type(16))) _Float16 v16h;
typedef __attribute__((ext_vector_type(8)))  _Float16 v8h;
typedef __attribute__((ext_vector_type(8)))  float    v8f;

#define DEVINL __device__ __forceinline__

namespace {
constexpr int kN = 4, kV = 2048, kC = 32, kH = 32, kT = 4;
constexpr int kNV   = kN * kV;          // 8192 nodes
constexpr int kKtot = 125;              // 5^3 spline kernels
constexpr int kS    = 8;                // 2^3 basis taps per edge
constexpr int kEper = 32768;
constexpr int kEtot = kN * kEper;       // 131072 edges
constexpr int kYld  = kKtot * kC;       // 4000 floats per node in Y
constexpr int kChunk = 16;              // spline kernels per LDS chunk (32 KB)
}

// ---------------- WMMA fragment helpers (ISA 7.12.2 layouts) ----------------

// A: 16x32 f16 tile, row-major in LDS (stride lda halfs). Lane L<16 holds row
// L, K = {k0+0..7, k0+16..23}; lane L>=16 holds row L-16, K = {k0+8..15, k0+24..31}.
DEVINL v16h load_a_frag(const _Float16* A, int lda, int k0, int lane) {
  const int row = lane & 15;
  const int kb  = (lane & 16) ? 8 : 0;
  const _Float16* p = A + row * lda + k0 + kb;
  union { v16h v; v8h h[2]; } u;
  u.h[0] = *(const v8h*)(p);
  u.h[1] = *(const v8h*)(p + 16);
  return u.v;
}

// B: 32x16 f16 tile given pre-transposed as Bt[N][K] row-major.
// Lane L<16 holds column L, K = {k0+0..7, k0+16..23}; lane L>=16 holds column
// L-16, K = {k0+8..15, k0+24..31} (same K order as A).
DEVINL v16h load_b_frag(const _Float16* Bt, int ldb, int k0, int lane) {
  const int col = lane & 15;
  const int kb  = (lane & 16) ? 8 : 0;
  const _Float16* p = Bt + col * ldb + k0 + kb;
  union { v16h v; v8h h[2]; } u;
  u.h[0] = *(const v8h*)(p);
  u.h[1] = *(const v8h*)(p + 16);
  return u.v;
}

DEVINL v8f wmma_16x16x32(v16h a, v16h b, v8f c) {
  // (neg_a, A, neg_b, B, c_mod, C, reuse_a, reuse_b)
  return __builtin_amdgcn_wmma_f32_16x16x32_f16(false, a, false, b, (short)0, c,
                                                false, false);
}

DEVINL v8f vzero8() {
  v8f z = {0.f, 0.f, 0.f, 0.f, 0.f, 0.f, 0.f, 0.f};
  return z;
}

DEVINL float act_elu(float x)     { return x > 0.f ? x : (__expf(x) - 1.f); }
DEVINL float act_sigmoid(float x) { return 1.f / (1.f + __expf(-x)); }

// C/D frag element i of lane L maps to (m = i + (L>=16)*8, n = L&15).
template <int NOUT>
DEVINL void frags_to_lds_f16(const v8f* f, _Float16* Alds, int lane) {
  const int n  = lane & 15;
  const int mh = (lane & 16) ? 8 : 0;
#pragma unroll
  for (int nt = 0; nt < NOUT / 16; ++nt)
#pragma unroll
    for (int i = 0; i < 8; ++i)
      Alds[(mh + i) * NOUT + nt * 16 + n] = (_Float16)f[nt][i];
}

template <int NOUT>
DEVINL void frags_store_f32(const v8f* f, float* out, int ldo, int lane) {
  const int n  = lane & 15;
  const int mh = (lane & 16) ? 8 : 0;
#pragma unroll
  for (int nt = 0; nt < NOUT / 16; ++nt)
#pragma unroll
    for (int i = 0; i < 8; ++i)
      out[(mh + i) * ldo + nt * 16 + n] = f[nt][i];
}

template <int NOUT>
DEVINL void frags_load_f32(v8f* f, const float* in, int ldo, int lane) {
  const int n  = lane & 15;
  const int mh = (lane & 16) ? 8 : 0;
#pragma unroll
  for (int nt = 0; nt < NOUT / 16; ++nt)
#pragma unroll
    for (int i = 0; i < 8; ++i)
      f[nt][i] = in[(mh + i) * ldo + nt * 16 + n];
}

// Stage 16 rows of f32 activations into LDS as f16 row-major [16,K].
template <int K>
DEVINL void stage_rows_f16(const float* X, int ldx, _Float16* Alds, int lane) {
  for (int idx = lane; idx < 16 * K; idx += 32) {
    const int r = idx / K, cc = idx % K;
    Alds[idx] = (_Float16)X[r * ldx + cc];
  }
}

// out[16,NOUT] = ACT(A[16,K] @ W^T + b); W pre-transposed to Bt[NOUT][K] f16.
// ACT: 0 = none, 1 = elu, 2 = tanh.
template <int K, int NOUT, int ACT>
DEVINL void dense_wave(const _Float16* Alds, const _Float16* BtW,
                       const float* bias, int lane, v8f* outf) {
  const int n = lane & 15;
#pragma unroll
  for (int nt = 0; nt < NOUT / 16; ++nt) {
    v8f c = vzero8();
#pragma unroll
    for (int kt = 0; kt < K / 32; ++kt) {
      v16h a = load_a_frag(Alds, K, kt * 32, lane);
      v16h b = load_b_frag(BtW + nt * 16 * K, K, kt * 32, lane);
      c = wmma_16x16x32(a, b, c);
    }
    const float bv = bias[nt * 16 + n];
#pragma unroll
    for (int i = 0; i < 8; ++i) {
      float xv = c[i] + bv;
      if (ACT == 1) xv = act_elu(xv);
      if (ACT == 2) xv = tanhf(xv);
      c[i] = xv;
    }
    outf[nt] = c;
  }
}

DEVINL void vaxpy2(v8f* o, const v8f* a, const v8f* b, float s) {
#pragma unroll
  for (int t = 0; t < 2; ++t)
#pragma unroll
    for (int i = 0; i < 8; ++i) o[t][i] = a[t][i] + s * b[t][i];
}

// ------------------------------ ODE / RK4 -----------------------------------

struct OdeWeights {
  const _Float16* in_w;  const float* in_b;   // [64,32] (already [N,K])
  const _Float16* l0_w;  const float* l0_b;   // [64,64]
  const _Float16* out_w; const float* out_b;  // [32,64]
};

DEVINL void ode_f_wave(const v8f hin[2], v8f fout[2], _Float16* a32,
                       _Float16* a64, _Float16* b64, const OdeWeights w,
                       int lane) {
  frags_to_lds_f16<32>(hin, a32, lane);
  v8f y1[4];
  dense_wave<32, 64, 1>(a32, w.in_w, w.in_b, lane, y1);
  frags_to_lds_f16<64>(y1, a64, lane);
  v8f y2[4];
  dense_wave<64, 64, 1>(a64, w.l0_w, w.l0_b, lane, y2);
  frags_to_lds_f16<64>(y2, b64, lane);
  dense_wave<64, 32, 2>(b64, w.out_w, w.out_b, lane, fout);
}

__global__ void __launch_bounds__(32)
rk4_kernel(const float* __restrict__ h, float* __restrict__ h_ode,
           OdeWeights w) {
  __shared__ _Float16 a32[16 * 32];
  __shared__ _Float16 a64[16 * 64];
  __shared__ _Float16 b64[16 * 64];
  const int lane = threadIdx.x;
  const int r0   = blockIdx.x * 16;
  v8f hc[2];
  frags_load_f32<32>(hc, h + r0 * kH, kH, lane);
  v8f k1[2], k2[2], k3[2], k4[2], tmp[2];
  ode_f_wave(hc, k1, a32, a64, b64, w, lane);
  vaxpy2(tmp, hc, k1, 0.5f);
  ode_f_wave(tmp, k2, a32, a64, b64, w, lane);
  vaxpy2(tmp, hc, k2, 0.5f);
  ode_f_wave(tmp, k3, a32, a64, b64, w, lane);
  vaxpy2(tmp, hc, k3, 1.0f);
  ode_f_wave(tmp, k4, a32, a64, b64, w, lane);
  v8f hn[2];
#pragma unroll
  for (int t = 0; t < 2; ++t)
#pragma unroll
    for (int i = 0; i < 8; ++i)
      hn[t][i] = hc[t][i] +
                 (k1[t][i] + 2.f * k2[t][i] + 2.f * k3[t][i] + k4[t][i]) *
                     (1.f / 6.f);
  frags_store_f32<32>(hn, h_ode + r0 * kH, kH, lane);
}

// ------------------------------ init h --------------------------------------

__global__ void __launch_bounds__(32)
init_h_kernel(const float* __restrict__ x, const _Float16* g1w,
              const float* g1b, const _Float16* g2w, const float* g2b,
              float* __restrict__ h) {
  __shared__ _Float16 a32[16 * 32];
  __shared__ _Float16 a64[16 * 64];
  const int lane = threadIdx.x;
  const int r0   = blockIdx.x * 16;
  for (int idx = lane; idx < 16 * 32; idx += 32) {
    const int r = idx >> 5, cc = idx & 31;
    // x layout [N,V,C,T]; node-major row (r0+r), channel cc, t = T-1.
    a32[idx] = (_Float16)x[(r0 + r) * (kC * kT) + cc * kT + (kT - 1)];
  }
  v8f y1[4];
  dense_wave<32, 64, 1>(a32, g1w, g1b, lane, y1);
  frags_to_lds_f16<64>(y1, a64, lane);
  v8f hf[2];
  dense_wave<64, 32, 2>(a64, g2w, g2b, lane, hf);
  frags_store_f32<32>(hf, h + r0 * kH, kH, lane);
}

// --------------------------- SplineConv: Y ----------------------------------
// Y[node, k, :] = X[node, :] @ w[k] for all 125 k.  8 waves per block, one
// 16-node tile per wave.  Spline weights are staged block-wide into LDS in
// 32 KB chunks with GLOBAL_LOAD_ASYNC_TO_LDS_B128 (ASYNCcnt), so each block
// reads the 256 KB weight set exactly once.

__global__ void __launch_bounds__(256)
spline_y_kernel(const float* __restrict__ X, const _Float16* __restrict__ wt,
                float* __restrict__ Y) {
  __shared__ _Float16 a32[8][16 * 32];          // 8 KB
  __shared__ _Float16 bbuf[kChunk * 1024];      // 32 KB weight chunk
  const int tid  = threadIdx.x;
  const int lane = tid & 31;
  const int wv   = tid >> 5;                    // wave 0..7
  const int r0   = blockIdx.x * 128 + wv * 16;  // 16-node tile per wave
  stage_rows_f16<32>(X + r0 * kC, kC, a32[wv], lane);
  const v16h a  = load_a_frag(a32[wv], 32, 0, lane);
  const int  n  = lane & 15;
  const int  mh = (lane & 16) ? 8 : 0;

  for (int base = 0; base < kKtot; base += kChunk) {
    const int nk = (kKtot - base) < kChunk ? (kKtot - base) : kChunk;
    // ---- async copy: nk * 1024 halfs = nk * 128 x 16-byte packets ----
    {
      const _Float16* gsrc = wt + base * 1024;
      const int nvec = nk * 128;
      for (int v = tid; v < nvec; v += 256) {
        const unsigned lds_off = (unsigned)(size_t)(bbuf + v * 8);
        const _Float16* gp = gsrc + v * 8;
        asm volatile("global_load_async_to_lds_b128 %0, %1, off"
                     :
                     : "v"(lds_off), "v"(gp)
                     : "memory");
      }
      asm volatile("s_wait_asynccnt 0x0" ::: "memory");
    }
    __syncthreads();
    // ---- compute nk kernels from LDS ----
    for (int kk = 0; kk < nk; ++kk) {
      const _Float16* Bt = bbuf + kk * 1024;    // [32 N][32 K] f16
      v16h b0 = load_b_frag(Bt, 32, 0, lane);
      v16h b1 = load_b_frag(Bt + 512, 32, 0, lane);
      v8f  c0 = wmma_16x16x32(a, b0, vzero8());
      v8f  c1 = wmma_16x16x32(a, b1, vzero8());
      const int k = base + kk;
#pragma unroll
      for (int i = 0; i < 8; ++i) {
        const int node = r0 + mh + i;
        float* yb = Y + node * kYld + k * kC;
        yb[n]      = c0[i];
        yb[16 + n] = c1[i];
      }
    }
    __syncthreads();                            // chunk buffer reuse
  }
}

// --------------------- SplineConv: gather + scatter --------------------------

__global__ void gather_scatter_kernel(const float* __restrict__ Y,
                                      const int* __restrict__ eidx,
                                      const float* __restrict__ basis,
                                      const int* __restrict__ widx,
                                      float* __restrict__ agg) {
  const int c = threadIdx.x;                              // channel 0..31
  const int e = blockIdx.x * blockDim.y + threadIdx.y;    // global edge
  if (e >= kEtot) return;
  const int j   = e & (kEper - 1);
  const int bn  = e >> 15;
  const int src = eidx[j] + bn * kV;
  const int dst = eidx[kEtot + j] + bn * kV;
  const float* yb = Y + src * kYld;
  float acc = 0.f;
#pragma unroll
  for (int s = 0; s < kS; ++s)
    acc += basis[j * kS + s] * yb[widx[j * kS + s] * kC + c];
  atomicAdd(&agg[dst * kC + c], acc);
}

// ------------------- SplineConv: finalize (mean + root + b) ------------------

__global__ void __launch_bounds__(32)
sconv_fin_kernel(const float* __restrict__ F, const float* __restrict__ agg,
                 const float* __restrict__ inv_deg,
                 const _Float16* __restrict__ root16,
                 const float* __restrict__ bias, float* __restrict__ out) {
  __shared__ _Float16 a32[16 * 32];
  const int lane = threadIdx.x;
  const int r0   = blockIdx.x * 16;
  stage_rows_f16<32>(F + r0 * kC, kC, a32, lane);
  const v16h a  = load_a_frag(a32, 32, 0, lane);
  v8f c0 = wmma_16x16x32(a, load_b_frag(root16, 32, 0, lane), vzero8());
  v8f c1 = wmma_16x16x32(a, load_b_frag(root16 + 512, 32, 0, lane), vzero8());
  const int n = lane & 15, mh = (lane & 16) ? 8 : 0;
#pragma unroll
  for (int i = 0; i < 8; ++i) {
    const int node = r0 + mh + i;
    const float id = inv_deg[node];
    out[node * kH + n]      = agg[node * kH + n] * id + c0[i] + bias[n];
    out[node * kH + 16 + n] = agg[node * kH + 16 + n] * id + c1[i] + bias[16 + n];
  }
}

// ----------------------------- small kernels --------------------------------

__global__ void spline_basis_kernel(const float* __restrict__ attr,
                                    float* __restrict__ basis,
                                    int* __restrict__ widx) {
  const int j = blockIdx.x * blockDim.x + threadIdx.x;
  if (j >= kEper) return;
  float lo[3], fr[3];
#pragma unroll
  for (int d = 0; d < 3; ++d) {
    const float u = attr[j * 3 + d] * 4.f;   // (KS-1)
    float l = floorf(u);
    l = fminf(fmaxf(l, 0.f), 3.f);           // clip to [0, KS-2]
    lo[d] = l;
    fr[d] = u - l;
  }
  const int strides[3] = {25, 5, 1};
#pragma unroll
  for (int s = 0; s < kS; ++s) {
    float w  = 1.f;
    int   id = 0;
#pragma unroll
    for (int d = 0; d < 3; ++d) {
      const int bit = (s >> (2 - d)) & 1;    // itertools.product bit order
      w  *= bit ? fr[d] : (1.f - fr[d]);
      id += ((int)lo[d] + bit) * strides[d];
    }
    basis[j * kS + s] = w;
    widx[j * kS + s]  = id;
  }
}

__global__ void deg_acc_kernel(const int* __restrict__ eidx,
                               float* __restrict__ deg) {
  const int e = blockIdx.x * blockDim.x + threadIdx.x;
  if (e >= kEtot) return;
  const int j   = e & (kEper - 1);
  const int dst = eidx[kEtot + j] + (e >> 15) * kV;
  atomicAdd(&deg[dst], 1.f);
}

__global__ void deg_fin_kernel(const float* __restrict__ deg,
                               float* __restrict__ invdeg, int n) {
  const int i = blockIdx.x * blockDim.x + threadIdx.x;
  if (i < n) invdeg[i] = 1.f / fmaxf(deg[i], 1.f);
}

__global__ void extract_x_kernel(const float* __restrict__ x,
                                 float* __restrict__ xstep, int t) {
  const int i = blockIdx.x * blockDim.x + threadIdx.x;
  if (i >= kNV * kC) return;
  const int node = i >> 5, c = i & 31;
  xstep[i] = x[node * (kC * kT) + c * kT + t];
}

__global__ void gru_combine_kernel(const float* __restrict__ sxr,
                                   const float* __restrict__ shr,
                                   const float* __restrict__ sxz,
                                   const float* __restrict__ shz,
                                   const float* __restrict__ sxn,
                                   const float* __restrict__ h_ode,
                                   float* __restrict__ h) {
  const int i = blockIdx.x * blockDim.x + threadIdx.x;
  if (i >= kNV * kH) return;
  const float hr = shr[i];
  const float r  = act_sigmoid(sxr[i] + hr);
  const float z  = act_sigmoid(sxz[i] + shz[i]);
  const float ng = tanhf(sxn[i] + r * hr);
  h[i] = (1.f - z) * ng + z * h_ode[i];
}

__global__ void zero_kernel(float* __restrict__ p, int n) {
  const int i = blockIdx.x * blockDim.x + threadIdx.x;
  if (i < n) p[i] = 0.f;
}

__global__ void copy_kernel(const float* __restrict__ in,
                            float* __restrict__ out, int n) {
  const int i = blockIdx.x * blockDim.x + threadIdx.x;
  if (i < n) out[i] = in[i];
}

__global__ void cvt_f16_kernel(const float* __restrict__ in,
                               _Float16* __restrict__ out, int n) {
  const int i = blockIdx.x * blockDim.x + threadIdx.x;
  if (i < n) out[i] = (_Float16)in[i];
}

// in: [b][K=32][N=32] -> out: [b][N=32][K=32] f16 (pre-transpose for B frags)
__global__ void cvt_t32_kernel(const float* __restrict__ in,
                               _Float16* __restrict__ out, int batches) {
  const int i = blockIdx.x * blockDim.x + threadIdx.x;
  if (i >= batches * 1024) return;
  const int b = i >> 10, r = i & 1023;
  const int nn = r >> 5, kk = r & 31;
  out[b * 1024 + nn * 32 + kk] = (_Float16)in[b * 1024 + kk * 32 + nn];
}

// ------------------------------- host side ----------------------------------

extern "C" void kernel_launch(void* const* d_in, const int* in_sizes, int n_in,
                              void* d_out, int out_size, void* d_ws,
                              size_t ws_size, hipStream_t stream) {
  (void)in_sizes; (void)out_size; (void)ws_size;
  const float* x = (const float*)d_in[0];
  // Params flattened depth-first in dict order:
  // 0 g1_w 1 g1_b 2 g2_w 3 g2_b | 4 in_w 5 in_b 6 l0_w 7 l0_b 8 out_w 9 out_b
  // 10..24: (w, root, b) for xr, hr, xz, hz, xn
  static const int psz[25] = {2048, 64, 2048, 32, 2048, 64, 4096, 64, 2048, 32,
                              128000, 1024, 32, 128000, 1024, 32,
                              128000, 1024, 32, 128000, 1024, 32,
                              128000, 1024, 32};
  const float* P[25];
  const int*   eidx;
  const float* eattr;
  if (n_in >= 28) {                     // each leaf is a separate input
    for (int i = 0; i < 25; ++i) P[i] = (const float*)d_in[1 + i];
    eidx  = (const int*)d_in[26];
    eattr = (const float*)d_in[27];
  } else if (n_in == 4) {               // params as one contiguous blob
    const float* blob = (const float*)d_in[1];
    size_t off = 0;
    for (int i = 0; i < 25; ++i) { P[i] = blob + off; off += psz[i]; }
    eidx  = (const int*)d_in[2];
    eattr = (const float*)d_in[3];
  } else {
    return;
  }

  // ---- carve workspace (256B aligned chunks) ----
  char* wp = (char*)d_ws;
  auto carve = [&](size_t bytes) -> void* {
    void* r = (void*)wp;
    wp += (bytes + 255) & ~(size_t)255;
    return r;
  };
  float* h      = (float*)carve((size_t)kNV * kH * 4);
  float* h_ode  = (float*)carve((size_t)kNV * kH * 4);
  float* xstep  = (float*)carve((size_t)kNV * kC * 4);
  float* basis  = (float*)carve((size_t)kEper * kS * 4);
  int*   widx   = (int*)carve((size_t)kEper * kS * 4);
  float* deg    = (float*)carve((size_t)kNV * 4);
  float* invdeg = (float*)carve((size_t)kNV * 4);
  float* agg    = (float*)carve((size_t)kNV * kH * 4);
  float* sc[5];
  for (int g = 0; g < 5; ++g) sc[g] = (float*)carve((size_t)kNV * kH * 4);
  _Float16* wt16[5];
  _Float16* rt16[5];
  for (int g = 0; g < 5; ++g) {
    wt16[g] = (_Float16*)carve((size_t)kKtot * 32 * 32 * 2);
    rt16[g] = (_Float16*)carve((size_t)32 * 32 * 2);
  }
  _Float16* g1w16  = (_Float16*)carve(2048 * 2);
  _Float16* g2w16  = (_Float16*)carve(2048 * 2);
  _Float16* inw16  = (_Float16*)carve(2048 * 2);
  _Float16* l0w16  = (_Float16*)carve(4096 * 2);
  _Float16* outw16 = (_Float16*)carve(2048 * 2);
  float* Y = (float*)carve((size_t)kNV * kYld * 4);   // 131 MB, reused per gate

  const int TB = 256;
  auto gsz = [](int n, int b) { return (n + b - 1) / b; };

  // ---- weight conversion / prepack ----
  cvt_f16_kernel<<<gsz(2048, TB), TB, 0, stream>>>(P[0], g1w16, 2048);
  cvt_f16_kernel<<<gsz(2048, TB), TB, 0, stream>>>(P[2], g2w16, 2048);
  cvt_f16_kernel<<<gsz(2048, TB), TB, 0, stream>>>(P[4], inw16, 2048);
  cvt_f16_kernel<<<gsz(4096, TB), TB, 0, stream>>>(P[6], l0w16, 4096);
  cvt_f16_kernel<<<gsz(2048, TB), TB, 0, stream>>>(P[8], outw16, 2048);
  for (int g = 0; g < 5; ++g) {
    cvt_t32_kernel<<<gsz(kKtot * 1024, TB), TB, 0, stream>>>(P[10 + 3 * g],
                                                             wt16[g], kKtot);
    cvt_t32_kernel<<<gsz(1024, TB), TB, 0, stream>>>(P[10 + 3 * g + 1],
                                                     rt16[g], 1);
  }

  // ---- graph preprocessing (once) ----
  spline_basis_kernel<<<gsz(kEper, TB), TB, 0, stream>>>(eattr, basis, widx);
  zero_kernel<<<gsz(kNV, TB), TB, 0, stream>>>(deg, kNV);
  deg_acc_kernel<<<gsz(kEtot, TB), TB, 0, stream>>>(eidx, deg);
  deg_fin_kernel<<<gsz(kNV, TB), TB, 0, stream>>>(deg, invdeg, kNV);

  // ---- initial hidden state ----
  init_h_kernel<<<kNV / 16, 32, 0, stream>>>(x, g1w16, P[1], g2w16, P[3], h);

  OdeWeights ow{inw16, P[5], l0w16, P[7], outw16, P[9]};

  // ---- reversed-time GRU scan ----
  for (int step = 0; step < kT; ++step) {
    const int t = kT - 1 - step;
    extract_x_kernel<<<gsz(kNV * kC, TB), TB, 0, stream>>>(x, xstep, t);
    rk4_kernel<<<kNV / 16, 32, 0, stream>>>(h, h_ode, ow);
    const float* feats[5] = {xstep, h_ode, xstep, h_ode, xstep};  // xr hr xz hz xn
    for (int g = 0; g < 5; ++g) {
      zero_kernel<<<gsz(kNV * kH, TB), TB, 0, stream>>>(agg, kNV * kH);
      spline_y_kernel<<<kNV / 128, 256, 0, stream>>>(feats[g], wt16[g], Y);
      gather_scatter_kernel<<<kEtot / 8, dim3(32, 8), 0, stream>>>(
          Y, eidx, basis, widx, agg);
      sconv_fin_kernel<<<kNV / 16, 32, 0, stream>>>(
          feats[g], agg, invdeg, rt16[g], P[10 + 3 * g + 2], sc[g]);
    }
    gru_combine_kernel<<<gsz(kNV * kH, TB), TB, 0, stream>>>(
        sc[0], sc[1], sc[2], sc[3], sc[4], h_ode, h);
  }

  copy_kernel<<<gsz(kNV * kH, TB), TB, 0, stream>>>(h, (float*)d_out, kNV * kH);
}
